// ldem_21139829031818
// MI455X (gfx1250) — matrix operand order repaired and checked by reading, not compiled, not run on hardware
//
#include <hip/hip_runtime.h>

// ---------------------------------------------------------------------------
// Problem constants (from reference: B=8, C=256, H=W=64, RATIO=4)
// ---------------------------------------------------------------------------
#define Bsz   8
#define Cdim  256
#define NTOK  4096           // 64*64 tokens per image
#define Mrow  (Bsz * NTOK)   // 32768 rows, main sequence
#define NTOKG 256            // 16*16 tokens, global branch
#define Mg    (Bsz * NTOKG)  // 2048 rows, global branch
#define NCH   32             // scan chunks (main), L = 128
#define NCHG  8              // scan chunks (global), L = 32

typedef __bf16 bf16_t;
typedef __attribute__((ext_vector_type(16))) __bf16 v16bf;
typedef __attribute__((ext_vector_type(8)))  float  v8f;

// ---------------------------------------------------------------------------
// Helpers
// ---------------------------------------------------------------------------
__device__ inline bf16_t f2bf(float f) {          // fp32 -> bf16 RNE
  union { float f; unsigned u; } v; v.f = f;
  unsigned r = v.u + 0x7FFFu + ((v.u >> 16) & 1u);
  unsigned short h = (unsigned short)(r >> 16);
  bf16_t b; __builtin_memcpy(&b, &h, 2); return b;
}
__device__ inline float sigm(float x) { return 1.f / (1.f + __expf(-x)); }

// Load a 32-byte WMMA fragment (16 bf16) as two 16B chunks `gap` elements apart.
__device__ inline v16bf ldfrag(const bf16_t* p, int gap) {
  union { uint4 q[2]; v16bf v; } u;
  u.q[0] = *reinterpret_cast<const uint4*>(p);
  u.q[1] = *reinterpret_cast<const uint4*>(p + gap);
  return u.v;
}

// ---------------------------------------------------------------------------
// Generic bf16 WMMA GEMM: C[M,Nc] = A[M,K] * B[K,Nc], f32 accumulate,
// fused epilogues selected by MODE. Wave tile 32x64 (2x4 accumulators),
// block = 4 waves -> 128x64. M%128==0, Nc%64==0, K%32==0 (all true here).
// ---------------------------------------------------------------------------
struct GP {
  const bf16_t* A; const bf16_t* Bp;
  int K, Nc, ntok, HW;
  const float *bias0, *bias1, *bias2;
  float *out0, *out1, *out2;
  const float *aux0, *aux1, *pos, *xres;
};

// MODE 0: out = acc + bias + pos[(row%ntok)*Nc+col]              (token proj)
// MODE 1: Nc=768 gate triple: a=exp(-softplus)=sigm(-x), g=sigm, o=sigm
// MODE 2: s=sigm(acc+bias); out = s*aux0 + (1-s)*aux1            (rho / eta)
// MODE 3: s=sigm(acc+bias); out = aux0 + s*aux1                  (lam)
// MODE 4: out = acc + bias                                       (v = vmap@Wl)
// MODE 5: transpose-store with residual: out[b,col,t] = x + acc + bias
template<int MODE>
__global__ __launch_bounds__(128) void gemm_bf16(GP p) {
  const int lane = threadIdx.x & 31;
  const int wave = threadIdx.x >> 5;
  const int m0   = blockIdx.x * 128 + wave * 32;
  const int n0   = blockIdx.y * 64;
  const int hl   = lane >> 4;      // K-half select for A, K-group for B
  const int lrow = lane & 15;

  v8f acc[2][4];
  for (int i = 0; i < 2; ++i)
    for (int j = 0; j < 4; ++j)
      for (int e = 0; e < 8; ++e) acc[i][j][e] = 0.f;

  const int nt16 = p.Nc >> 4;
  const bf16_t* a0p = p.A + (size_t)(m0 + lrow) * p.K + hl * 8;
  const bf16_t* a1p = a0p + (size_t)16 * p.K;
  const int ktn = p.K >> 5;

  for (int kt = 0; kt < ktn; ++kt) {
    // A fragments: lane holds row m, K = hl*8 + [0..7] and 16 + hl*8 + [0..7]
    v16bf af0 = ldfrag(a0p + kt * 32, 16);
    v16bf af1 = ldfrag(a1p + kt * 32, 16);
    // prefetch A one k-tile ahead (streams from L2/HBM)
    if (kt + 1 < ktn) {
      __builtin_prefetch(a0p + (kt + 1) * 32, 0, 1);
      __builtin_prefetch(a1p + (kt + 1) * 32, 0, 1);
    }
    // B fragments: pre-packed so each lane reads 32 contiguous bytes
    const bf16_t* bp = p.Bp + (((size_t)kt * nt16 + (n0 >> 4)) * 32 + lane) * 16;
    v16bf b0 = ldfrag(bp,        8);
    v16bf b1 = ldfrag(bp +  512, 8);
    v16bf b2 = ldfrag(bp + 1024, 8);
    v16bf b3 = ldfrag(bp + 1536, 8);

    acc[0][0] = __builtin_amdgcn_wmma_f32_16x16x32_bf16(false, af0, false, b0, (short)0, acc[0][0], false, false);
    acc[0][1] = __builtin_amdgcn_wmma_f32_16x16x32_bf16(false, af0, false, b1, (short)0, acc[0][1], false, false);
    acc[0][2] = __builtin_amdgcn_wmma_f32_16x16x32_bf16(false, af0, false, b2, (short)0, acc[0][2], false, false);
    acc[0][3] = __builtin_amdgcn_wmma_f32_16x16x32_bf16(false, af0, false, b3, (short)0, acc[0][3], false, false);
    acc[1][0] = __builtin_amdgcn_wmma_f32_16x16x32_bf16(false, af1, false, b0, (short)0, acc[1][0], false, false);
    acc[1][1] = __builtin_amdgcn_wmma_f32_16x16x32_bf16(false, af1, false, b1, (short)0, acc[1][1], false, false);
    acc[1][2] = __builtin_amdgcn_wmma_f32_16x16x32_bf16(false, af1, false, b2, (short)0, acc[1][2], false, false);
    acc[1][3] = __builtin_amdgcn_wmma_f32_16x16x32_bf16(false, af1, false, b3, (short)0, acc[1][3], false, false);
  }

  // Epilogue. C/D layout: lane L, VGPR r -> row = r + (L/16)*8, col = L%16.
  for (int mi = 0; mi < 2; ++mi)
    for (int ni = 0; ni < 4; ++ni)
#pragma unroll
      for (int r = 0; r < 8; ++r) {
        int row = m0 + mi * 16 + hl * 8 + r;
        int col = n0 + ni * 16 + lrow;
        float val = acc[mi][ni][r];
        if constexpr (MODE == 0) {
          size_t o = (size_t)row * p.Nc + col;
          p.out0[o] = val + p.bias0[col] + p.pos[(size_t)(row % p.ntok) * p.Nc + col];
        } else if constexpr (MODE == 1) {
          int sec = col >> 8, cc = col & 255;
          const float* bs = sec == 0 ? p.bias0 : (sec == 1 ? p.bias1 : p.bias2);
          float xg = val + bs[cc];
          size_t o = (size_t)row * 256 + cc;
          if (sec == 0)      p.out0[o] = 1.f / (1.f + __expf(xg)); // exp(-softplus(x))
          else if (sec == 1) p.out1[o] = sigm(xg);
          else               p.out2[o] = sigm(xg);
        } else if constexpr (MODE == 2) {
          size_t o = (size_t)row * p.Nc + col;
          float s = sigm(val + p.bias0[col]);
          p.out0[o] = s * p.aux0[o] + (1.f - s) * p.aux1[o];
        } else if constexpr (MODE == 3) {
          size_t o = (size_t)row * p.Nc + col;
          float s = sigm(val + p.bias0[col]);
          p.out0[o] = p.aux0[o] + s * p.aux1[o];
        } else if constexpr (MODE == 4) {
          size_t o = (size_t)row * p.Nc + col;
          p.out0[o] = val + p.bias0[col];
        } else { // MODE 5: residual + transpose to [B,C,H*W]
          int b = row / p.HW, t = row - b * p.HW;
          size_t o = ((size_t)(b * p.Nc + col)) * p.HW + t;
          p.out0[o] = p.xres[o] + val + p.bias0[col];
        }
      }
}

// ---------------------------------------------------------------------------
// Weight repack: W[k, n] -> B-fragment order so each lane's v16bf is
// contiguous: off = ((kt*nt16 + n/16)*32 + lane)*16 + j,
// lane = ((k%32 / 8)&1)*16 + n%16, j = k%8 + (k%32 / 16)*8.
// ---------------------------------------------------------------------------
__device__ inline void packw_store(bf16_t* dst, int k, int n, int Nc, float v) {
  int kt = k >> 5, kk = k & 31;
  int lane = (((kk >> 3) & 1) << 4) | (n & 15);
  int j = (kk & 7) | ((kk >> 4) << 3);
  size_t off = (((size_t)kt * (Nc >> 4) + (n >> 4)) * 32 + lane) * 16 + j;
  dst[off] = f2bf(v);
}

__global__ void pack_w(const float* __restrict__ W, bf16_t* __restrict__ dst, int K, int Nc) {
  int idx = blockIdx.x * blockDim.x + threadIdx.x;
  if (idx >= K * Nc) return;
  int k = idx / Nc, n = idx - k * Nc;
  packw_store(dst, k, n, Nc, W[idx]);
}

// Concat [Wf|Ww|Wo] along output dim -> [512, 768] packed
__global__ void pack_w3(const float* __restrict__ W0, const float* __restrict__ W1,
                        const float* __restrict__ W2, bf16_t* __restrict__ dst, int K) {
  const int Nc = 768;
  int idx = blockIdx.x * blockDim.x + threadIdx.x;
  if (idx >= K * Nc) return;
  int k = idx / Nc, n = idx - k * Nc;
  const float* src = n < 256 ? W0 : (n < 512 ? W1 : W2);
  packw_store(dst, k, n, Nc, src[k * 256 + (n & 255)]);
}

// ---------------------------------------------------------------------------
// Elementwise / packing kernels (c-fastest layouts, coalesced stores)
// ---------------------------------------------------------------------------
__global__ void pack_seq(const float* __restrict__ x, bf16_t* __restrict__ dst, int total) {
  int idx = blockIdx.x * blockDim.x + threadIdx.x;
  if (idx >= total) return;
  int c = idx & 255; int m = idx >> 8;
  int b = m >> 12; int t = m & 4095;
  dst[idx] = f2bf(x[((size_t)(b * 256 + c)) * 4096 + t]);
}

__global__ void pack_cat_ctx(const float* __restrict__ u, const float* __restrict__ ctx,
                             bf16_t* __restrict__ dst, int ntok, int total) {
  int idx = blockIdx.x * blockDim.x + threadIdx.x;
  if (idx >= total) return;
  int j = idx & 511; int m = idx >> 9;
  int b = m / ntok;
  float v = (j < 256) ? u[(size_t)m * 256 + j] : ctx[b * 256 + (j - 256)];
  dst[idx] = f2bf(v);
}

__global__ void pack_cat3(const float* __restrict__ p0, const float* __restrict__ p1,
                          const float* __restrict__ p2, bf16_t* __restrict__ dst, int total) {
  int idx = blockIdx.x * blockDim.x + threadIdx.x;
  if (idx >= total) return;
  int j = idx % 768; int m = idx / 768;
  int sec = j >> 8;
  const float* s = sec == 0 ? p0 : (sec == 1 ? p1 : p2);
  dst[idx] = f2bf(s[(size_t)m * 256 + (j & 255)]);
}

__global__ void pack_cat2(const float* __restrict__ p0, const float* __restrict__ p1,
                          bf16_t* __restrict__ dst, int total) {
  int idx = blockIdx.x * blockDim.x + threadIdx.x;
  if (idx >= total) return;
  int j = idx & 511; int m = idx >> 9;
  const float* s = j < 256 ? p0 : p1;
  dst[idx] = f2bf(s[(size_t)m * 256 + (j & 255)]);
}

__global__ void pack1(const float* __restrict__ src, bf16_t* __restrict__ dst, int total) {
  int idx = blockIdx.x * blockDim.x + threadIdx.x;
  if (idx >= total) return;
  dst[idx] = f2bf(src[idx]);
}

// Pooled-LN context: ctx[b,c] = LN_c(mean_n u[b,n,c]) * g + beta
__global__ void context_ln(const float* __restrict__ u, const float* __restrict__ g,
                           const float* __restrict__ beta, float* __restrict__ ctx, int ntok) {
  int b = blockIdx.x, c = threadIdx.x;
  const float* up = u + ((size_t)b * ntok) * 256 + c;
  float s = 0.f;
  for (int n = 0; n < ntok; ++n) s += up[(size_t)n * 256];
  float m = s / (float)ntok;
  __shared__ float r1[256], r2[256];
  r1[c] = m; r2[c] = m * m;
  __syncthreads();
  for (int st = 128; st > 0; st >>= 1) {
    if (c < st) { r1[c] += r1[c + st]; r2[c] += r2[c + st]; }
    __syncthreads();
  }
  float mu  = r1[0] * (1.f / 256.f);
  float var = r2[0] * (1.f / 256.f) - mu * mu;
  ctx[b * 256 + c] = (m - mu) * rsqrtf(var + 1e-5f) * g[c] + beta[c];
}

// jax.image.resize (bilinear, half-pixel, antialiased triangle when downsampling)
__global__ void resize_pos(const float* __restrict__ pos, float* __restrict__ out,
                           int H, int W, int total) {
  int idx = blockIdx.x * blockDim.x + threadIdx.x;
  if (idx >= total) return;
  int c = idx & 255; int t = idx >> 8; int w = t % W; int h = t / W;
  const int S = 32;
  float scy = (float)H / S, scx = (float)W / S;
  float fy = (h + 0.5f) / scy - 0.5f;
  float fx = (w + 0.5f) / scx - 0.5f;
  float ky = scy < 1.f ? scy : 1.f;
  float kx = scx < 1.f ? scx : 1.f;
  int y0 = (int)floorf(fy) - 3, x0 = (int)floorf(fx) - 3;
  float acc = 0.f, wsum = 0.f;
  for (int py = y0; py < y0 + 8; ++py) {
    float wy = 1.f - fabsf(py - fy) * ky;
    if (wy <= 0.f) continue;
    int cy = min(max(py, 0), S - 1);
    for (int px = x0; px < x0 + 8; ++px) {
      float wx = 1.f - fabsf(px - fx) * kx;
      if (wx <= 0.f) continue;
      int cx = min(max(px, 0), S - 1);
      float wt = wy * wx;
      acc += wt * pos[c * 1024 + cy * 32 + cx];
      wsum += wt;
    }
  }
  out[(size_t)t * 256 + c] = acc / wsum;
}

// 4x4 exact avg-pool + transpose + bf16 pack: x[B,C,64,64] -> seq_g[Mg,256]
__global__ void pool_pack(const float* __restrict__ x, bf16_t* __restrict__ dst, int total) {
  int idx = blockIdx.x * blockDim.x + threadIdx.x;
  if (idx >= total) return;
  int c = idx & 255; int m = idx >> 8;
  int b = m >> 8; int t = m & 255;
  int gh = t >> 4, gw = t & 15;
  const float* xp = x + ((size_t)(b * 256 + c)) * 4096 + (gh * 4) * 64 + gw * 4;
  float s = 0.f;
  for (int dy = 0; dy < 4; ++dy)
    for (int dx = 0; dx < 4; ++dx) s += xp[dy * 64 + dx];
  dst[idx] = f2bf(s * 0.0625f);
}

// y_g [B,16,16,C] -> bilinear x4 upsample -> [B,4096,C]
__global__ void upsample_yg(const float* __restrict__ yg, float* __restrict__ out, int total) {
  int idx = blockIdx.x * blockDim.x + threadIdx.x;
  if (idx >= total) return;
  int c = idx & 255; int m = idx >> 8; int b = m >> 12; int t = m & 4095;
  int h = t >> 6, w = t & 63;
  float fy = (h + 0.5f) * 0.25f - 0.5f;
  float fx = (w + 0.5f) * 0.25f - 0.5f;
  int y0i = (int)floorf(fy); float ay = fy - y0i;
  int x0i = (int)floorf(fx); float ax = fx - x0i;
  int yA = min(max(y0i, 0), 15), yB = min(max(y0i + 1, 0), 15);
  int xA = min(max(x0i, 0), 15), xB = min(max(x0i + 1, 0), 15);
  const float* yp = yg + (size_t)b * 256 * 256 + c;
  float vAA = yp[(size_t)(yA * 16 + xA) * 256];
  float vAB = yp[(size_t)(yA * 16 + xB) * 256];
  float vBA = yp[(size_t)(yB * 16 + xA) * 256];
  float vBB = yp[(size_t)(yB * 16 + xB) * 256];
  out[idx] = (1.f - ay) * ((1.f - ax) * vAA + ax * vAB) + ay * ((1.f - ax) * vBA + ax * vBB);
}

// Depthwise 3x3 SAME conv, transposed bf16 pack for the Wl GEMM
__global__ void dwconv_pack(const float* __restrict__ x, const float* __restrict__ wgt,
                            const float* __restrict__ bias, bf16_t* __restrict__ dst, int total) {
  int idx = blockIdx.x * blockDim.x + threadIdx.x;
  if (idx >= total) return;
  int t = idx & 4095; int bc = idx >> 12;
  int c = bc & 255; int b = bc >> 8;
  int h = t >> 6, w = t & 63;
  const float* xp = x + (size_t)bc * 4096;
  const float* wp = wgt + c * 9;
  float acc = bias[c];
#pragma unroll
  for (int dy = -1; dy <= 1; ++dy) {
    int hh = h + dy; if (hh < 0 || hh > 63) continue;
#pragma unroll
    for (int dx = -1; dx <= 1; ++dx) {
      int ww = w + dx; if (ww < 0 || ww > 63) continue;
      acc += xp[hh * 64 + ww] * wp[(dy + 1) * 3 + (dx + 1)];
    }
  }
  dst[((size_t)((b << 12) | t)) * 256 + c] = f2bf(acc);
}

// ---------------------------------------------------------------------------
// Chunked gated linear scan (3 passes). s = a*s + (1-a)*g*u; y = o*s + (1-o)*u
// dir=0 forward, dir=1 backward (gates are per-token => shared across dirs).
// ---------------------------------------------------------------------------
__global__ void scan_pass1(const float* __restrict__ ga, const float* __restrict__ gg,
                           const float* __restrict__ u, float* __restrict__ cA,
                           float* __restrict__ cS, int ntok, int nchunk, int dir, int total) {
  int idx = blockIdx.x * blockDim.x + threadIdx.x;
  if (idx >= total) return;
  int c = idx & 255; int rest = idx >> 8;
  int ch = rest % nchunk; int b = rest / nchunk;
  int L = ntok / nchunk;
  float A = 1.f, S = 0.f;
  for (int i = 0; i < L; ++i) {
    int n = dir ? (ntok - 1 - ch * L - i) : (ch * L + i);
    size_t off = ((size_t)b * ntok + n) * 256 + c;
    float a = ga[off];
    S = a * S + (1.f - a) * gg[off] * u[off];
    A *= a;
  }
  cA[idx] = A; cS[idx] = S;
}

__global__ void scan_pass2(const float* __restrict__ cA, const float* __restrict__ cS,
                           float* __restrict__ s0, int nchunk, int total) {
  int idx = blockIdx.x * blockDim.x + threadIdx.x;
  if (idx >= total) return;
  int c = idx & 255; int b = idx >> 8;
  float s = 0.f;
  for (int ch = 0; ch < nchunk; ++ch) {
    size_t off = ((size_t)(b * nchunk + ch)) * 256 + c;
    s0[off] = s;
    s = cA[off] * s + cS[off];
  }
}

__global__ void scan_pass3(const float* __restrict__ ga, const float* __restrict__ gg,
                           const float* __restrict__ go, const float* __restrict__ u,
                           const float* __restrict__ s0, float* __restrict__ y,
                           int ntok, int nchunk, int dir, int total) {
  int idx = blockIdx.x * blockDim.x + threadIdx.x;
  if (idx >= total) return;
  int c = idx & 255; int rest = idx >> 8;
  int ch = rest % nchunk; int b = rest / nchunk;
  int L = ntok / nchunk;
  float s = s0[idx];
  for (int i = 0; i < L; ++i) {
    int n = dir ? (ntok - 1 - ch * L - i) : (ch * L + i);
    size_t off = ((size_t)b * ntok + n) * 256 + c;
    float a = ga[off];
    float uu = u[off];
    s = a * s + (1.f - a) * gg[off] * uu;
    float o = go[off];
    y[off] = o * s + (1.f - o) * uu;
  }
}

// ---------------------------------------------------------------------------
// Host orchestration
// ---------------------------------------------------------------------------
extern "C" void kernel_launch(void* const* d_in, const int* in_sizes, int n_in,
                              void* d_out, int out_size, void* d_ws, size_t ws_size,
                              hipStream_t stream) {
  (void)in_sizes; (void)n_in; (void)out_size; (void)ws_size;

  const float* x    = (const float*)d_in[0];
  const float* Wt   = (const float*)d_in[1];
  const float* bt   = (const float*)d_in[2];
  const float* posf = (const float*)d_in[3];
  const float* posg = (const float*)d_in[4];
  const float* lng  = (const float*)d_in[5];
  const float* lnb  = (const float*)d_in[6];
  const float* Wf_  = (const float*)d_in[7];
  const float* bf_  = (const float*)d_in[8];
  const float* Ww_  = (const float*)d_in[9];
  const float* bw_  = (const float*)d_in[10];
  const float* Wo_  = (const float*)d_in[11];
  const float* bo_  = (const float*)d_in[12];
  const float* Wr_  = (const float*)d_in[13];
  const float* br_  = (const float*)d_in[14];
  const float* Wgi_ = (const float*)d_in[15];
  const float* bgi_ = (const float*)d_in[16];
  const float* dww  = (const float*)d_in[17];
  const float* dwb  = (const float*)d_in[18];
  const float* Wl_  = (const float*)d_in[19];
  const float* bl_  = (const float*)d_in[20];
  const float* Wlf_ = (const float*)d_in[21];
  const float* blf_ = (const float*)d_in[22];
  const float* Wout_= (const float*)d_in[23];
  const float* bout_= (const float*)d_in[24];
  float* outp = (float*)d_out;

  // Bump allocator over d_ws
  char* base = (char*)d_ws;
  size_t off = 0;
  auto alloc = [&](size_t bytes) -> void* {
    void* p = base + off;
    off += (bytes + 255) & ~(size_t)255;
    return p;
  };

  bf16_t* WtP   = (bf16_t*)alloc((size_t)256 * 256 * 2);
  bf16_t* WgP   = (bf16_t*)alloc((size_t)512 * 768 * 2);
  bf16_t* WrP   = (bf16_t*)alloc((size_t)768 * 256 * 2);
  bf16_t* WgiP  = (bf16_t*)alloc((size_t)768 * 256 * 2);
  bf16_t* WlP   = (bf16_t*)alloc((size_t)256 * 256 * 2);
  bf16_t* WlfP  = (bf16_t*)alloc((size_t)512 * 256 * 2);
  bf16_t* WoutP = (bf16_t*)alloc((size_t)256 * 256 * 2);
  float*  posF  = (float*)alloc((size_t)NTOK  * 256 * 4);
  float*  posG  = (float*)alloc((size_t)NTOKG * 256 * 4);
  float*  u     = (float*)alloc((size_t)Mrow * 256 * 4);
  bf16_t* cat   = (bf16_t*)alloc((size_t)Mrow * 768 * 2);
  float*  gA    = (float*)alloc((size_t)Mrow * 256 * 4);
  float*  gG    = (float*)alloc((size_t)Mrow * 256 * 4);
  float*  gO    = (float*)alloc((size_t)Mrow * 256 * 4);
  float*  yf    = (float*)alloc((size_t)Mrow * 256 * 4);
  float*  yb    = (float*)alloc((size_t)Mrow * 256 * 4);
  float*  cAr   = (float*)alloc((size_t)Bsz * NCH * 256 * 4);
  float*  cSr   = (float*)alloc((size_t)Bsz * NCH * 256 * 4);
  float*  s0b   = (float*)alloc((size_t)Bsz * NCH * 256 * 4);
  float*  ctx   = (float*)alloc((size_t)Bsz * 256 * 4);
  float*  ctxg  = (float*)alloc((size_t)Bsz * 256 * 4);
  float*  ug    = (float*)alloc((size_t)Mg * 256 * 4);
  bf16_t* gbuf  = (bf16_t*)alloc((size_t)Mg * 768 * 2);
  float*  ggA   = (float*)alloc((size_t)Mg * 256 * 4);
  float*  ggG   = (float*)alloc((size_t)Mg * 256 * 4);
  float*  ggO   = (float*)alloc((size_t)Mg * 256 * 4);
  float*  yg    = (float*)alloc((size_t)Mg * 256 * 4);

  // Aliases (lifetimes do not overlap):
  float* y    = gA;   // rho output
  float* ygup = gG;   // upsampled global branch
  float* z    = gO;   // lam output
  float* v    = yf;   // depthwise+Wl output (after rho consumed yf)
  float* uout = yb;   // eta output (after rho consumed yb)

  const int TB = 256;
  auto blk = [](int n, int t) { return dim3((unsigned)((n + t - 1) / t)); };

  // 1. Repack weights to WMMA B-fragment order (bf16)
  pack_w <<<blk(256*256, TB), TB, 0, stream>>>(Wt,  WtP,  256, 256);
  pack_w3<<<blk(512*768, TB), TB, 0, stream>>>(Wf_, Ww_, Wo_, WgP, 512);
  pack_w <<<blk(768*256, TB), TB, 0, stream>>>(Wr_,  WrP,  768, 256);
  pack_w <<<blk(768*256, TB), TB, 0, stream>>>(Wgi_, WgiP, 768, 256);
  pack_w <<<blk(256*256, TB), TB, 0, stream>>>(Wl_,  WlP,  256, 256);
  pack_w <<<blk(512*256, TB), TB, 0, stream>>>(Wlf_, WlfP, 512, 256);
  pack_w <<<blk(256*256, TB), TB, 0, stream>>>(Wout_, WoutP, 256, 256);

  // 2. Positional embedding tables
  resize_pos<<<blk(NTOK*256,  TB), TB, 0, stream>>>(posf, posF, 64, 64, NTOK*256);
  resize_pos<<<blk(NTOKG*256, TB), TB, 0, stream>>>(posg, posG, 16, 16, NTOKG*256);

  // 3. u = seq @ Wt + bt + pos_fine  (seq staged bf16 in `cat`)
  pack_seq<<<blk(Mrow*256, TB), TB, 0, stream>>>(x, cat, Mrow*256);
  {
    GP p{}; p.A = cat; p.Bp = WtP; p.K = 256; p.Nc = 256; p.ntok = NTOK;
    p.bias0 = bt; p.out0 = u; p.pos = posF;
    gemm_bf16<0><<<dim3(Mrow/128, 4), 128, 0, stream>>>(p);
  }

  // 4. Pooled LN context, gate GEMM (a = sigm(-x), g, o) — shared by fwd & bwd scans
  context_ln<<<Bsz, 256, 0, stream>>>(u, lng, lnb, ctx, NTOK);
  pack_cat_ctx<<<blk(Mrow*512, TB), TB, 0, stream>>>(u, ctx, cat, NTOK, Mrow*512);
  {
    GP p{}; p.A = cat; p.Bp = WgP; p.K = 512; p.Nc = 768;
    p.bias0 = bf_; p.bias1 = bw_; p.bias2 = bo_;
    p.out0 = gA; p.out1 = gG; p.out2 = gO;
    gemm_bf16<1><<<dim3(Mrow/128, 12), 128, 0, stream>>>(p);
  }

  // 5. Bidirectional chunked scans
  const int T1 = Bsz*NCH*256, T2 = Bsz*256;
  scan_pass1<<<blk(T1, TB), TB, 0, stream>>>(gA, gG, u, cAr, cSr, NTOK, NCH, 0, T1);
  scan_pass2<<<blk(T2, TB), TB, 0, stream>>>(cAr, cSr, s0b, NCH, T2);
  scan_pass3<<<blk(T1, TB), TB, 0, stream>>>(gA, gG, gO, u, s0b, yf, NTOK, NCH, 0, T1);
  scan_pass1<<<blk(T1, TB), TB, 0, stream>>>(gA, gG, u, cAr, cSr, NTOK, NCH, 1, T1);
  scan_pass2<<<blk(T2, TB), TB, 0, stream>>>(cAr, cSr, s0b, NCH, T2);
  scan_pass3<<<blk(T1, TB), TB, 0, stream>>>(gA, gG, gO, u, s0b, yb, NTOK, NCH, 1, T1);

  // 6. Global branch: pool -> proj -> context -> gates -> scan -> upsample
  pool_pack<<<blk(Mg*256, TB), TB, 0, stream>>>(x, gbuf, Mg*256);
  {
    GP p{}; p.A = gbuf; p.Bp = WtP; p.K = 256; p.Nc = 256; p.ntok = NTOKG;
    p.bias0 = bt; p.out0 = ug; p.pos = posG;
    gemm_bf16<0><<<dim3(Mg/128, 4), 128, 0, stream>>>(p);
  }
  context_ln<<<Bsz, 256, 0, stream>>>(ug, lng, lnb, ctxg, NTOKG);
  pack_cat_ctx<<<blk(Mg*512, TB), TB, 0, stream>>>(ug, ctxg, gbuf, NTOKG, Mg*512);
  {
    GP p{}; p.A = gbuf; p.Bp = WgP; p.K = 512; p.Nc = 768;
    p.bias0 = bf_; p.bias1 = bw_; p.bias2 = bo_;
    p.out0 = ggA; p.out1 = ggG; p.out2 = ggO;
    gemm_bf16<1><<<dim3(Mg/128, 12), 128, 0, stream>>>(p);
  }
  const int G1 = Bsz*NCHG*256;
  scan_pass1<<<blk(G1, TB), TB, 0, stream>>>(ggA, ggG, ug, cAr, cSr, NTOKG, NCHG, 0, G1);
  scan_pass2<<<blk(T2, TB), TB, 0, stream>>>(cAr, cSr, s0b, NCHG, T2);
  scan_pass3<<<blk(G1, TB), TB, 0, stream>>>(ggA, ggG, ggO, ug, s0b, yg, NTOKG, NCHG, 0, G1);
  upsample_yg<<<blk(Mrow*256, TB), TB, 0, stream>>>(yg, ygup, Mrow*256);

  // 7. rho gate: y = rho*yf + (1-rho)*yb   (gA freed by now -> holds y)
  pack_cat3<<<blk(Mrow*768, TB), TB, 0, stream>>>(u, yf, yb, cat, Mrow*768);
  {
    GP p{}; p.A = cat; p.Bp = WrP; p.K = 768; p.Nc = 256;
    p.bias0 = br_; p.aux0 = yf; p.aux1 = yb; p.out0 = y;
    gemm_bf16<2><<<dim3(Mrow/128, 4), 128, 0, stream>>>(p);
  }

  // 8. lam gate: z = y + lam*ygup
  pack_cat3<<<blk(Mrow*768, TB), TB, 0, stream>>>(y, ygup, u, cat, Mrow*768);
  {
    GP p{}; p.A = cat; p.Bp = WgiP; p.K = 768; p.Nc = 256;
    p.bias0 = bgi_; p.aux0 = y; p.aux1 = ygup; p.out0 = z;
    gemm_bf16<3><<<dim3(Mrow/128, 4), 128, 0, stream>>>(p);
  }

  // 9. Local branch: depthwise conv -> v = vmap @ Wl + bl
  dwconv_pack<<<blk(Mrow*256, TB), TB, 0, stream>>>(x, dww, dwb, cat, Mrow*256);
  {
    GP p{}; p.A = cat; p.Bp = WlP; p.K = 256; p.Nc = 256;
    p.bias0 = bl_; p.out0 = v;
    gemm_bf16<4><<<dim3(Mrow/128, 4), 128, 0, stream>>>(p);
  }

  // 10. eta gate: u_out = eta*v + (1-eta)*z
  pack_cat2<<<blk(Mrow*512, TB), TB, 0, stream>>>(v, z, cat, Mrow*512);
  {
    GP p{}; p.A = cat; p.Bp = WlfP; p.K = 512; p.Nc = 256;
    p.bias0 = blf_; p.aux0 = v; p.aux1 = z; p.out0 = uout;
    gemm_bf16<2><<<dim3(Mrow/128, 4), 128, 0, stream>>>(p);
  }

  // 11. Output projection + residual, transposed store to [B,C,H,W]
  pack1<<<blk(Mrow*256, TB), TB, 0, stream>>>(uout, cat, Mrow*256);
  {
    GP p{}; p.A = cat; p.Bp = WoutP; p.K = 256; p.Nc = 256; p.HW = NTOK;
    p.bias0 = bout_; p.xres = x; p.out0 = outp;
    gemm_bf16<5><<<dim3(Mrow/128, 4), 128, 0, stream>>>(p);
  }
}
